// DAGLinkPredictor_26963804684290
// MI455X (gfx1250) — matrix operand order repaired and checked by reading, not compiled, not run on hardware
//
#include <hip/hip_runtime.h>
#include <hip/hip_bf16.h>

typedef __bf16 bf16_t;
typedef __attribute__((ext_vector_type(16))) __bf16 v16bf;
typedef __attribute__((ext_vector_type(8)))  __bf16 v8bf;
typedef __attribute__((ext_vector_type(8)))  float  v8f;

#define EMBD 16

// ---------- helpers ----------
__device__ __forceinline__ unsigned fenc(float f) {
  unsigned u = __float_as_uint(f);
  return (u >> 31) ? ~u : (u | 0x80000000u);
}
__device__ __forceinline__ float fdec(unsigned u) {
  u = (u >> 31) ? (u & 0x7fffffffu) : ~u;
  return __uint_as_float(u);
}

// ---------- node prep: types + bf16 [N,64] (emb[type] ++ feats, zero pad 48->64) ----------
__global__ void k_prep_nodes(const float* __restrict__ x, const float* __restrict__ emb,
                             int* __restrict__ types, bf16_t* __restrict__ xc, int N) {
  int tid = blockIdx.x * blockDim.x + threadIdx.x;
  int n = tid >> 6;
  int j = tid & 63;
  if (n >= N) return;
  int t = (int)x[(size_t)n * 33];
  if (j == 0) types[n] = t;
  float v;
  if (j < EMBD)    v = emb[(size_t)t * EMBD + j];
  else if (j < 48) v = x[(size_t)n * 33 + 1 + (j - EMBD)];
  else             v = 0.f;
  xc[(size_t)n * 64 + j] = (bf16_t)v;
}

// ---------- W [K,Nc] f32 -> Wt [Nc,Kpad] bf16 (zero pad K) ----------
__global__ void k_transpose_pad(const float* __restrict__ W, bf16_t* __restrict__ Wt,
                                int K, int Nc, int Kpad) {
  int tid = blockIdx.x * blockDim.x + threadIdx.x;
  if (tid >= Nc * Kpad) return;
  int c = tid / Kpad, k = tid % Kpad;
  Wt[(size_t)c * Kpad + k] = (k < K) ? (bf16_t)W[(size_t)k * Nc + c] : (bf16_t)0.f;
}

// ---------- WMMA GEMM: C[M,Nc] = A[M,K](bf16,row) * Bt[Nc,K](bf16,row-of-transposed) ----------
// one wave per 16x16 tile, 4 waves/block along N
__global__ void k_gemm_wmma(const bf16_t* __restrict__ A, const bf16_t* __restrict__ Bt,
                            float* __restrict__ C, int M, int K, int Nc,
                            const float* __restrict__ bias, int relu) {
  int lane = threadIdx.x & 31;
  int wave = threadIdx.x >> 5;
  int half = lane >> 4;     // 0: lanes 0-15, 1: lanes 16-31
  int l16  = lane & 15;
  int m0 = blockIdx.x * 16;
  int c0 = (blockIdx.y * 4 + wave) * 16;
  if (m0 >= M || c0 >= Nc) return;   // wave-uniform
  const bf16_t* aptr = A  + (size_t)(m0 + l16) * K + half * 16;
  const bf16_t* bptr = Bt + (size_t)(c0 + l16) * K + half * 16;
  v8f acc = {};
  for (int kt = 0; kt < K; kt += 32) {
    union { v16bf v; v8bf h[2]; } a;
    a.h[0] = *(const v8bf*)(aptr + kt);        // K: kt + half*16 .. +7
    a.h[1] = *(const v8bf*)(aptr + kt + 8);    // K: kt+8+half*16 .. +7
    v16bf b = *(const v16bf*)(bptr + kt);      // col c0+l16, K: kt+half*16 .. +15
    acc = __builtin_amdgcn_wmma_f32_16x16x32_bf16(false, a.v, false, b,
                                                  (short)0, acc, false, false);
  }
  int col = c0 + l16;
  int mb = m0 + half * 8;
  float bv = bias ? bias[col] : 0.f;
#pragma unroll
  for (int r = 0; r < 8; ++r) {
    float v = acc[r] + bv;
    if (relu) v = v > 0.f ? v : 0.f;
    C[(size_t)(mb + r) * Nc + col] = v;
  }
}

// ---------- per-(node,head) attention coefficients + init max/denom ----------
__global__ void k_attn_init(const float* __restrict__ h, const float* __restrict__ asrc,
                            const float* __restrict__ adst, float* __restrict__ als,
                            float* __restrict__ ald, unsigned* __restrict__ menc,
                            float* __restrict__ den, int N, int H, int D) {
  int tid = blockIdx.x * blockDim.x + threadIdx.x;
  if (tid >= N * H) return;
  int n = tid / H, hh = tid - n * H;
  const float* row = h + (size_t)n * H * D + (size_t)hh * D;
  const float* as = asrc + hh * D;
  const float* ad = adst + hh * D;
  float s = 0.f, d = 0.f;
  for (int i = 0; i < D; ++i) { float v = row[i]; s += v * as[i]; d += v * ad[i]; }
  als[tid] = s; ald[tid] = d; menc[tid] = 0u; den[tid] = 0.f;
}

__global__ void k_zero(float* __restrict__ p, size_t n) {
  size_t tid = (size_t)blockIdx.x * blockDim.x + threadIdx.x;
  if (tid < n) p[tid] = 0.f;
}

// ---------- edge pass 1: logits + segment max (via ordered-uint atomicMax) ----------
__global__ void k_edge_max(const int* __restrict__ ei, int E, int EP,
                           const float* __restrict__ als, const float* __restrict__ ald,
                           unsigned* __restrict__ menc, float* __restrict__ ev, int hsh) {
  int tid = blockIdx.x * blockDim.x + threadIdx.x;
  int e = tid >> hsh;
  if (e >= EP) return;
  int H = 1 << hsh;
  int h = tid & (H - 1);
  int s, d;
  if (e < E) { s = ei[e]; d = ei[E + e]; } else { s = d = e - E; }
  float v = als[(size_t)s * H + h] + ald[(size_t)d * H + h];
  v = v > 0.f ? v : 0.2f * v;      // leaky_relu 0.2
  ev[(size_t)e * H + h] = v;
  atomicMax(menc + (size_t)d * H + h, fenc(v));
}

// ---------- edge pass 2: exp(logit - max), accumulate denom ----------
__global__ void k_edge_exp(const int* __restrict__ ei, int E, int EP,
                           const unsigned* __restrict__ menc, float* __restrict__ den,
                           float* __restrict__ ev, int hsh) {
  int tid = blockIdx.x * blockDim.x + threadIdx.x;
  int e = tid >> hsh;
  if (e >= EP) return;
  int H = 1 << hsh;
  int h = tid & (H - 1);
  int d;
  if (e < E) { d = ei[E + e]; } else { d = e - E; }
  float m = fdec(menc[(size_t)d * H + h]);
  float v = __expf(ev[(size_t)e * H + h] - m);
  ev[(size_t)e * H + h] = v;
  atomicAdd(den + (size_t)d * H + h, v);
}

// ---------- edge pass 3: alpha-weighted message aggregation (block per edge) ----------
__global__ void k_edge_agg(const int* __restrict__ ei, int E,
                           const float* __restrict__ hfeat, const float* __restrict__ ev,
                           const float* __restrict__ den, float* __restrict__ agg,
                           int hsh, int dsh, int F) {
  int e = blockIdx.x;
  int j = threadIdx.x;
  int s, d;
  if (e < E) { s = ei[e]; d = ei[E + e]; } else { s = d = e - E; }
  int H = 1 << hsh;
  int h = j >> dsh;
  float alpha = ev[(size_t)e * H + h] / (den[(size_t)d * H + h] + 1e-16f);
  atomicAdd(agg + (size_t)d * F + j, hfeat[(size_t)s * F + j] * alpha);
}

// ---------- bias + optional ELU -> bf16 activations for next GEMM ----------
__global__ void k_post(const float* __restrict__ agg, const float* __restrict__ bias,
                       bf16_t* __restrict__ outbf, size_t total, int Fmask, int elu) {
  size_t tid = (size_t)blockIdx.x * blockDim.x + threadIdx.x;
  if (tid >= total) return;
  int j = (int)(tid & (size_t)Fmask);
  float v = agg[tid] + bias[j];
  if (elu && v < 0.f) v = expm1f(v);
  outbf[tid] = (bf16_t)v;
}

// ---------- decoder: gather ef = concat(z[ls], z[ld]) as bf16 [EL,256] ----------
__global__ void k_gather_ef(const bf16_t* __restrict__ zbf, const int* __restrict__ eli,
                            int EL, bf16_t* __restrict__ ef) {
  int tid = blockIdx.x * blockDim.x + threadIdx.x;
  if (tid >= EL * 256) return;
  int l = tid >> 8;
  int f = tid & 255;
  int node = (f < 128) ? eli[l] : eli[EL + l];
  ef[tid] = zbf[(size_t)node * 128 + (f & 127)];
}

// ---------- final: hd[EL,64] @ Wl2 + bl2 + trans_bias lookup ----------
__global__ void k_final(const float* __restrict__ hd, const float* __restrict__ Wl2,
                        const float* __restrict__ bl2, const float* __restrict__ tb,
                        const int* __restrict__ types, const int* __restrict__ eli,
                        float* __restrict__ out, int EL, int NT) {
  int l = blockIdx.x * blockDim.x + threadIdx.x;
  if (l >= EL) return;
  const float* row = hd + (size_t)l * 64;
  float s = 0.f;
#pragma unroll
  for (int j = 0; j < 64; ++j) s += row[j] * Wl2[j];
  int ts = types[eli[l]];
  int td = types[eli[EL + l]];
  out[l] = s + bl2[0] + tb[(size_t)ts * NT + td];
}

extern "C" void kernel_launch(void* const* d_in, const int* in_sizes, int n_in,
                              void* d_out, int out_size, void* d_ws, size_t ws_size,
                              hipStream_t stream) {
  const float* x    = (const float*)d_in[0];
  const int*   ei   = (const int*)d_in[1];
  const int*   eli  = (const int*)d_in[2];
  const float* emb  = (const float*)d_in[3];
  const float* W1   = (const float*)d_in[4];
  const float* as1  = (const float*)d_in[5];
  const float* ad1  = (const float*)d_in[6];
  const float* b1   = (const float*)d_in[7];
  const float* W2   = (const float*)d_in[8];
  const float* as2  = (const float*)d_in[9];
  const float* ad2  = (const float*)d_in[10];
  const float* b2   = (const float*)d_in[11];
  const float* W3   = (const float*)d_in[12];
  const float* as3  = (const float*)d_in[13];
  const float* ad3  = (const float*)d_in[14];
  const float* b3   = (const float*)d_in[15];
  const float* Wl1  = (const float*)d_in[16];
  const float* bl1  = (const float*)d_in[17];
  const float* Wl2  = (const float*)d_in[18];
  const float* bl2  = (const float*)d_in[19];
  const float* tb   = (const float*)d_in[20];
  float* out = (float*)d_out;

  const int N  = in_sizes[0] / 33;
  const int E  = in_sizes[1] / 2;
  const int EL = in_sizes[2] / 2;
  const int NT = in_sizes[3] / EMBD;
  const int EP = E + N;

  // workspace layout
  char* p = (char*)d_ws;
  auto alloc = [&](size_t b) -> void* {
    void* r = (void*)p;
    p += (b + 255) & ~(size_t)255;
    return r;
  };
  int*      types = (int*)alloc((size_t)N * 4);
  bf16_t*   xc    = (bf16_t*)alloc((size_t)N * 64 * 2);
  bf16_t*   act1  = (bf16_t*)alloc((size_t)N * 256 * 2);
  bf16_t*   act2  = (bf16_t*)alloc((size_t)N * 256 * 2);
  bf16_t*   zbf   = (bf16_t*)alloc((size_t)N * 128 * 2);
  bf16_t*   W1t   = (bf16_t*)alloc((size_t)256 * 64 * 2);
  bf16_t*   W2t   = (bf16_t*)alloc((size_t)256 * 256 * 2);
  bf16_t*   W3t   = (bf16_t*)alloc((size_t)128 * 256 * 2);
  bf16_t*   Wl1t  = (bf16_t*)alloc((size_t)64 * 256 * 2);
  float*    hbuf  = (float*)alloc((size_t)N * 256 * 4);
  float*    agg   = (float*)alloc((size_t)N * 256 * 4);
  float*    als   = (float*)alloc((size_t)N * 4 * 4);
  float*    ald   = (float*)alloc((size_t)N * 4 * 4);
  unsigned* menc  = (unsigned*)alloc((size_t)N * 4 * 4);
  float*    den   = (float*)alloc((size_t)N * 4 * 4);
  float*    ev    = (float*)alloc((size_t)EP * 4 * 4);
  bf16_t*   efbf  = (bf16_t*)hbuf;  // alias: EL*256*2 == N*256*4 bytes, lifetimes disjoint
  float*    hd    = agg;            // alias: EL*64*4 <= N*256*4, lifetimes disjoint

  const int T = 256;

  // prep
  k_prep_nodes<<<(N * 64 + T - 1) / T, T, 0, stream>>>(x, emb, types, xc, N);
  k_transpose_pad<<<(256 * 64 + T - 1) / T, T, 0, stream>>>(W1, W1t, 48, 256, 64);
  k_transpose_pad<<<(256 * 256 + T - 1) / T, T, 0, stream>>>(W2, W2t, 256, 256, 256);
  k_transpose_pad<<<(128 * 256 + T - 1) / T, T, 0, stream>>>(W3, W3t, 256, 128, 256);
  k_transpose_pad<<<(64 * 256 + T - 1) / T, T, 0, stream>>>(Wl1, Wl1t, 256, 64, 256);

  auto run_layer = [&](const bf16_t* Abf, int Kpad, const bf16_t* Wt,
                       const float* asc, const float* adc, const float* bias,
                       int H, int hsh, int D, int dsh, int elu, bf16_t* outbf) {
    int F = H * D;
    dim3 g(N / 16, F / 64);
    k_gemm_wmma<<<g, 128, 0, stream>>>(Abf, Wt, hbuf, N, Kpad, F, nullptr, 0);
    k_attn_init<<<(N * H + T - 1) / T, T, 0, stream>>>(hbuf, asc, adc, als, ald, menc, den, N, H, D);
    size_t tot = (size_t)N * F;
    k_zero<<<(unsigned)((tot + T - 1) / T), T, 0, stream>>>(agg, tot);
    int eth = EP * H;
    k_edge_max<<<(eth + T - 1) / T, T, 0, stream>>>(ei, E, EP, als, ald, menc, ev, hsh);
    k_edge_exp<<<(eth + T - 1) / T, T, 0, stream>>>(ei, E, EP, menc, den, ev, hsh);
    k_edge_agg<<<EP, F, 0, stream>>>(ei, E, hbuf, ev, den, agg, hsh, dsh, F);
    k_post<<<(unsigned)((tot + T - 1) / T), T, 0, stream>>>(agg, bias, outbf, tot, F - 1, elu);
  };

  // GAT layers: (heads, D) = (4,64) -> (2,128) -> (1,128)
  run_layer(xc,   64,  W1t, as1, ad1, b1, 4, 2, 64, 6, 1, act1);
  run_layer(act1, 256, W2t, as2, ad2, b2, 2, 1, 128, 7, 1, act2);
  run_layer(act2, 256, W3t, as3, ad3, b3, 1, 0, 128, 7, 0, zbf);

  // decoder
  k_gather_ef<<<(EL * 256 + T - 1) / T, T, 0, stream>>>(zbf, eli, EL, efbf);
  dim3 gd(EL / 16, 1);
  k_gemm_wmma<<<gd, 128, 0, stream>>>(efbf, Wl1t, hd, EL, 256, 64, bl1, 1);
  k_final<<<(EL + T - 1) / T, T, 0, stream>>>(hd, Wl2, bl2, tb, types, eli, out, EL, NT);
}
